// CausalSelfAttention_15607911153923
// MI455X (gfx1250) — matrix operand order repaired and checked by reading, not compiled, not run on hardware
//
#include <hip/hip_runtime.h>

// ---------------------------------------------------------------------------
// Causal self-attention fwd for MI455X (gfx1250, wave32, WMMA).
//   D_MODEL=1024, N_HEADS=16, HEAD_DIM=64, B=4, T=2048.
// All GEMM work routed through v_wmma_f32_16x16x32_bf16, register
// double-buffered so WMMAs overlap outstanding loads. Attention additionally
// drives the Tensor Data Mover (tensor_load_to_lds) to prefetch the next
// K-tile into LDS concurrently with QK WMMAs + softmax.
// ---------------------------------------------------------------------------

typedef __bf16 bf16;
typedef bf16  v16bf __attribute__((ext_vector_type(16)));
typedef bf16  v8bf  __attribute__((ext_vector_type(8)));
typedef float v8f   __attribute__((ext_vector_type(8)));
typedef unsigned u32x4 __attribute__((ext_vector_type(4)));
typedef int      i32x4 __attribute__((ext_vector_type(4)));
typedef int      i32x8 __attribute__((ext_vector_type(8)));

#define DM   1024
#define HEADS 16
#define HD   64
#define BATCH 4
#define SEQ  2048
#define ROWS (BATCH * SEQ)      // 8192

static __device__ __forceinline__ bf16 f2bf(float f) {
    union { float f; unsigned u; } c; c.f = f;
    unsigned u = c.u;
    u += 0x7fffu + ((u >> 16) & 1u);       // round-to-nearest-even
    unsigned short h = (unsigned short)(u >> 16);
    return __builtin_bit_cast(bf16, h);
}

// A-fragment (16x32 bf16, row-major source, leading dim lda elements).
// Lane m=lane&15 holds row m; half h=lane>>4 picks K chunks [h*8,h*8+8) and
// [16+h*8, 16+h*8+8)  (ISA 7.12.2, 16-bit A 16x32).
static __device__ __forceinline__ v16bf load_a_frag(const bf16* base, int lda, int lane) {
    int m = lane & 15, h = lane >> 4;
    const bf16* p = base + (size_t)m * lda + h * 8;
    v8bf lo = *(const v8bf*)p;
    v8bf hi = *(const v8bf*)(p + 16);
    return __builtin_shufflevector(lo, hi, 0,1,2,3,4,5,6,7,8,9,10,11,12,13,14,15);
}

// B-fragment (32x16 bf16): source is B^T row-major [N][K] (== B column-major).
// Lane n=lane&15 holds column n; K chunks [h*8,h*8+8) and [16+h*8,16+h*8+8).
static __device__ __forceinline__ v16bf load_b_frag(const bf16* baseT, int ldb, int lane) {
    int n = lane & 15, h = lane >> 4;
    const bf16* p = baseT + (size_t)n * ldb + h * 8;
    v8bf lo = *(const v8bf*)p;
    v8bf hi = *(const v8bf*)(p + 16);
    return __builtin_shufflevector(lo, hi, 0,1,2,3,4,5,6,7,8,9,10,11,12,13,14,15);
}

static __device__ __forceinline__ v8f wmma_bf16(v16bf a, v16bf b, v8f c) {
    return __builtin_amdgcn_wmma_f32_16x16x32_bf16(false, a, false, b, (short)0, c, false, false);
}

// ---------------------------------------------------------------------------
// TDM: stage a [rows x 64] bf16 tile (row-major, stride 64) into LDS.
// D# per CDNA5 ISA (2-D tensor, groups 2/3 zero).
// This toolchain exposes the 6-arg builtin form.
// ---------------------------------------------------------------------------
static __device__ __forceinline__ void tdm_load_k_tile(const bf16* gsrc,
                                                       unsigned lds_off,
                                                       unsigned rows) {
    unsigned long long ga = (unsigned long long)(const void*)gsrc;
    u32x4 g0;
    g0[0] = 1u;                                              // count=1, user mode
    g0[1] = lds_off;                                         // lds_addr (bytes)
    g0[2] = (unsigned)ga;                                    // global_addr[31:0]
    g0[3] = (unsigned)((ga >> 32) & 0x01FFFFFFu) | (2u << 30); // addr[56:32] | type=2

    i32x8 g1;
    g1[0] = (int)(1u << 16);          // workgroup_mask=0 | data_size=1 (2B)
    g1[1] = (int)(64u << 16);         // tensor_dim0[15:0]=64 (in elements)
    g1[2] = (int)((unsigned)SEQ << 16); // tensor_dim0[31:16]=0 | tensor_dim1[15:0]=2048
    g1[3] = (int)(64u << 16);         // tensor_dim1[31:16]=0 | tile_dim0=64
    g1[4] = (int)rows;                // tile_dim1=rows | tile_dim2=0
    g1[5] = 64;                       // tensor_dim0_stride[31:0]=64
    g1[6] = 0;                        // stride0[47:32]=0 | dim1_stride[15:0]=0 (2-D)
    g1[7] = 0;
    i32x4 gz4 = {0, 0, 0, 0};
    i32x8 gz8 = {0, 0, 0, 0, 0, 0, 0, 0};
    __builtin_amdgcn_tensor_load_to_lds(g0, g1, gz4, gz4, gz8, 0);
}

// ---------------------------------------------------------------------------
// Converters (one-shot, memory-bound prep)
// ---------------------------------------------------------------------------
__global__ void cvt_f32_bf16(const float* __restrict__ in, bf16* __restrict__ out, int n) {
    int i = blockIdx.x * blockDim.x + threadIdx.x;
    if (i < n) out[i] = f2bf(in[i]);
}

// in[k][n] (rows x cols) -> out[n][k]  (bf16, transposed)
__global__ void cvt_transpose(const float* __restrict__ in, bf16* __restrict__ out,
                              int rows, int cols) {
    int i = blockIdx.x * blockDim.x + threadIdx.x;
    if (i < rows * cols) {
        int n = i / rows, k = i % rows;
        out[(size_t)n * rows + k] = f2bf(in[(size_t)k * cols + n]);
    }
}

// ---------------------------------------------------------------------------
// Double-buffered GEMM core: wave computes 32(m) x 64(n), K = DM.
// acc[mt*4+t] accumulates (m0+mt*16, n0+t*16) 16x16 tile.
// ---------------------------------------------------------------------------
static __device__ __forceinline__ void gemm_core(const bf16* __restrict__ A,
                                                 const bf16* __restrict__ Wt,
                                                 int m0, int n0, int lane, v8f acc[8]) {
    v16bf a0 = load_a_frag(A + (size_t)m0 * DM,        DM, lane);
    v16bf a1 = load_a_frag(A + (size_t)(m0 + 16) * DM, DM, lane);
    v16bf b0 = load_b_frag(Wt + (size_t)(n0 +  0) * DM, DM, lane);
    v16bf b1 = load_b_frag(Wt + (size_t)(n0 + 16) * DM, DM, lane);
    v16bf b2 = load_b_frag(Wt + (size_t)(n0 + 32) * DM, DM, lane);
    v16bf b3 = load_b_frag(Wt + (size_t)(n0 + 48) * DM, DM, lane);

    for (int k0 = 0; k0 < DM; k0 += 32) {
        // branchless prefetch of next k-step (wraps to k=0 on the final,
        // dead iteration so no control flow / no OOB)
        int kn = (k0 + 32) & (DM - 1);
        v16bf na0 = load_a_frag(A + (size_t)m0 * DM + kn,        DM, lane);
        v16bf na1 = load_a_frag(A + (size_t)(m0 + 16) * DM + kn, DM, lane);
        v16bf nb0 = load_b_frag(Wt + (size_t)(n0 +  0) * DM + kn, DM, lane);
        v16bf nb1 = load_b_frag(Wt + (size_t)(n0 + 16) * DM + kn, DM, lane);
        v16bf nb2 = load_b_frag(Wt + (size_t)(n0 + 32) * DM + kn, DM, lane);
        v16bf nb3 = load_b_frag(Wt + (size_t)(n0 + 48) * DM + kn, DM, lane);

        acc[0] = wmma_bf16(a0, b0, acc[0]);
        acc[1] = wmma_bf16(a0, b1, acc[1]);
        acc[2] = wmma_bf16(a0, b2, acc[2]);
        acc[3] = wmma_bf16(a0, b3, acc[3]);
        acc[4] = wmma_bf16(a1, b0, acc[4]);
        acc[5] = wmma_bf16(a1, b1, acc[5]);
        acc[6] = wmma_bf16(a1, b2, acc[6]);
        acc[7] = wmma_bf16(a1, b3, acc[7]);

        a0 = na0; a1 = na1; b0 = nb0; b1 = nb1; b2 = nb2; b3 = nb3;
    }
}

// ---------------------------------------------------------------------------
// QKV GEMM: [8192,1024] x [1024,3072] + bias.  Block = 4 waves, 128x64 tile.
// Scatters into Q[B,H,T,D], K[B,H,T,D], Vt[B,H,D,T] as bf16.
// ---------------------------------------------------------------------------
__global__ __launch_bounds__(128) void qkv_gemm(const bf16* __restrict__ X,
                                                const bf16* __restrict__ Wt,
                                                const float* __restrict__ bias,
                                                bf16* __restrict__ Qb,
                                                bf16* __restrict__ Kb,
                                                bf16* __restrict__ Vt) {
    int lane = threadIdx.x & 31, wv = threadIdx.x >> 5;
    int m0 = blockIdx.x * 128 + wv * 32;
    int n0 = blockIdx.y * 64;

    v8f acc[8] = {{}, {}, {}, {}, {}, {}, {}, {}};
    gemm_core(X, Wt, m0, n0, lane, acc);

    int col = lane & 15, half = lane >> 4;
    int sec = n0 >> 10;                       // 0=Q 1=K 2=V (uniform: 64 | 1024)
#pragma unroll
    for (int mt = 0; mt < 2; mt++) {
#pragma unroll
        for (int t = 0; t < 4; t++) {
#pragma unroll
            for (int r = 0; r < 8; r++) {
                int row  = m0 + mt * 16 + half * 8 + r;   // global token row
                int colg = n0 + t * 16 + col;             // global qkv column
                float v  = acc[mt * 4 + t][r] + bias[colg];
                int cc = colg & 1023, head = cc >> 6, d = cc & 63;
                int b = row >> 11, tt = row & 2047;
                int bh = b * HEADS + head;
                if (sec == 0)      Qb[((size_t)bh * SEQ + tt) * HD + d] = f2bf(v);
                else if (sec == 1) Kb[((size_t)bh * SEQ + tt) * HD + d] = f2bf(v);
                else               Vt[((size_t)bh * HD + d) * SEQ + tt] = f2bf(v);
            }
        }
    }
}

// ---------------------------------------------------------------------------
// Flash attention: 1 wave per 16-query tile, online softmax, causal.
// S = Q K^T via 2 WMMAs per 16-key tile; P->A-layout via LDS; O += P V.
// V fragments load together with K fragments (overlap QK WMMAs + softmax),
// and the TDM prefetches the next chunk's K-tile into LDS in the background.
// ---------------------------------------------------------------------------
__global__ __launch_bounds__(32) void attn_kernel(const bf16* __restrict__ Q,
                                                  const bf16* __restrict__ K,
                                                  const bf16* __restrict__ Vt,
                                                  bf16* __restrict__ Obf) {
    __shared__ bf16 pbuf[16 * 32];
    __shared__ bf16 kstage[32 * HD];          // TDM staging tile (4 KB)
    int lane = threadIdx.x & 31;
    int col = lane & 15, half = lane >> 4;

    int blk = blockIdx.x;            // ((b*H + h)*128 + qt)
    int qt = blk & 127, bh = blk >> 7;
    int q0 = qt * 16;

    const bf16* Qb = Q  + (size_t)bh * SEQ * HD;
    const bf16* Kb = K  + (size_t)bh * SEQ * HD;
    const bf16* Vb = Vt + (size_t)bh * HD * SEQ;

    // LDS byte offset of the staging tile (flat LDS aperture: addr[31:0])
    unsigned kstage_off = (unsigned)(size_t)(void*)&kstage[0];

    v16bf aq0 = load_a_frag(Qb + (size_t)q0 * HD + 0,  HD, lane);
    v16bf aq1 = load_a_frag(Qb + (size_t)q0 * HD + 32, HD, lane);

    v8f o0 = {}, o1 = {}, o2 = {}, o3 = {};
    float mr[8], lr[8];
#pragma unroll
    for (int r = 0; r < 8; r++) { mr[r] = -3.0e38f; lr[r] = 0.f; }

    const int qhi = q0 + 15;
    for (int j0 = 0; j0 <= qhi; j0 += 32) {
        // ---- TDM background prefetch of the next K chunk into LDS
        int j0n = (j0 + 32) & (SEQ - 1);      // wraps harmlessly on last chunk
        __builtin_amdgcn_s_wait_tensorcnt(3); // throttle outstanding TDM ops
        tdm_load_k_tile(Kb + (size_t)j0n * HD, kstage_off, 32);

        // ---- issue ALL global loads for this chunk up front
        v16bf bk00 = load_b_frag(Kb + (size_t)(j0)      * HD + 0,  HD, lane);
        v16bf bk01 = load_b_frag(Kb + (size_t)(j0)      * HD + 32, HD, lane);
        v16bf bk10 = load_b_frag(Kb + (size_t)(j0 + 16) * HD + 0,  HD, lane);
        v16bf bk11 = load_b_frag(Kb + (size_t)(j0 + 16) * HD + 32, HD, lane);
        v16bf bv0  = load_b_frag(Vb + (size_t)0  * SEQ + j0, SEQ, lane);
        v16bf bv1  = load_b_frag(Vb + (size_t)16 * SEQ + j0, SEQ, lane);
        v16bf bv2  = load_b_frag(Vb + (size_t)32 * SEQ + j0, SEQ, lane);
        v16bf bv3  = load_b_frag(Vb + (size_t)48 * SEQ + j0, SEQ, lane);

        // ---- S = Q K^T (scale 1/8) over 32 keys (two 16-wide tiles)
        v8f s0 = {}, s1 = {};
        s0 = wmma_bf16(aq0, bk00, s0); s0 = wmma_bf16(aq1, bk01, s0);
        s1 = wmma_bf16(aq0, bk10, s1); s1 = wmma_bf16(aq1, bk11, s1);

        int j_0 = j0 + col, j_1 = j0 + 16 + col;
#pragma unroll
        for (int r = 0; r < 8; r++) {
            int qrow = q0 + half * 8 + r;
            float x0 = s0[r] * 0.125f; if (j_0 > qrow) x0 = -3.0e38f;
            float x1 = s1[r] * 0.125f; if (j_1 > qrow) x1 = -3.0e38f;
            // row max over the 16 lanes of this half (same matrix row)
            float rm = fmaxf(x0, x1);
            rm = fmaxf(rm, __shfl_xor(rm, 1, 32));
            rm = fmaxf(rm, __shfl_xor(rm, 2, 32));
            rm = fmaxf(rm, __shfl_xor(rm, 4, 32));
            rm = fmaxf(rm, __shfl_xor(rm, 8, 32));
            float mn   = fmaxf(mr[r], rm);
            float corr = __expf(mr[r] - mn);
            float p0 = __expf(x0 - mn), p1 = __expf(x1 - mn);
            float rs = p0 + p1;
            rs += __shfl_xor(rs, 1, 32);
            rs += __shfl_xor(rs, 2, 32);
            rs += __shfl_xor(rs, 4, 32);
            rs += __shfl_xor(rs, 8, 32);
            lr[r] = lr[r] * corr + rs;
            mr[r] = mn;
            o0[r] *= corr; o1[r] *= corr; o2[r] *= corr; o3[r] *= corr;
            pbuf[(half * 8 + r) * 32 + col]      = f2bf(p0);
            pbuf[(half * 8 + r) * 32 + 16 + col] = f2bf(p1);
        }
        __syncthreads();
        v16bf pA = load_a_frag(&pbuf[0], 32, lane);   // C-layout -> A-layout via LDS

        // ---- O += P V  (V stored transposed: column-major B, contiguous)
        o0 = wmma_bf16(pA, bv0, o0);
        o1 = wmma_bf16(pA, bv1, o1);
        o2 = wmma_bf16(pA, bv2, o2);
        o3 = wmma_bf16(pA, bv3, o3);
        __syncthreads();
    }
    __builtin_amdgcn_s_wait_tensorcnt(0);     // drain TDM before reusing LDS

    // epilogue: attn_out[b, t, h, d]  (row-major [B*T, 1024] for proj GEMM)
    int b = bh >> 4, h = bh & 15;
#pragma unroll
    for (int r = 0; r < 8; r++) {
        int qrow = q0 + half * 8 + r;
        float inv = 1.0f / lr[r];
        size_t base = (((size_t)b * SEQ + qrow) * HEADS + h) * HD;
        Obf[base +  0 + col] = f2bf(o0[r] * inv);
        Obf[base + 16 + col] = f2bf(o1[r] * inv);
        Obf[base + 32 + col] = f2bf(o2[r] * inv);
        Obf[base + 48 + col] = f2bf(o3[r] * inv);
    }
}

// ---------------------------------------------------------------------------
// Output projection: [8192,1024] x [1024,1024] + bias -> f32 d_out
// ---------------------------------------------------------------------------
__global__ __launch_bounds__(128) void proj_gemm(const bf16* __restrict__ A,
                                                 const bf16* __restrict__ Wt,
                                                 const float* __restrict__ bias,
                                                 float* __restrict__ out) {
    int lane = threadIdx.x & 31, wv = threadIdx.x >> 5;
    int m0 = blockIdx.x * 128 + wv * 32;
    int n0 = blockIdx.y * 64;

    v8f acc[8] = {{}, {}, {}, {}, {}, {}, {}, {}};
    gemm_core(A, Wt, m0, n0, lane, acc);

    int col = lane & 15, half = lane >> 4;
#pragma unroll
    for (int mt = 0; mt < 2; mt++) {
#pragma unroll
        for (int t = 0; t < 4; t++) {
#pragma unroll
            for (int r = 0; r < 8; r++) {
                int row  = m0 + mt * 16 + half * 8 + r;
                int colg = n0 + t * 16 + col;
                out[(size_t)row * DM + colg] = acc[mt * 4 + t][r] + bias[colg];
            }
        }
    }
}

// ---------------------------------------------------------------------------
extern "C" void kernel_launch(void* const* d_in, const int* in_sizes, int n_in,
                              void* d_out, int out_size, void* d_ws, size_t ws_size,
                              hipStream_t stream) {
    const float* x      = (const float*)d_in[0];   // [4,2048,1024]
    const float* w_qkv  = (const float*)d_in[1];   // [1024,3072]
    const float* b_qkv  = (const float*)d_in[2];   // [3072]
    const float* w_proj = (const float*)d_in[3];   // [1024,1024]
    const float* b_proj = (const float*)d_in[4];   // [1024]
    float* out = (float*)d_out;

    char* ws = (char*)d_ws;
    const size_t SZ_X   = (size_t)ROWS * DM * 2;        // 16.8 MB (bf16)
    const size_t SZ_WQ  = (size_t)3 * DM * DM * 2;      // 6.3 MB
    const size_t SZ_WP  = (size_t)DM * DM * 2;          // 2.1 MB
    const size_t SZ_QKV = (size_t)ROWS * DM * 2;        // per tensor

    bf16* xbf    = (bf16*)(ws);                                   // dead after QKV gemm
    bf16* wqkvT  = (bf16*)(ws + SZ_X);
    bf16* wprojT = (bf16*)(ws + SZ_X + SZ_WQ);
    bf16* Qb     = (bf16*)(ws + SZ_X + SZ_WQ + SZ_WP);
    bf16* Kb     = (bf16*)(ws + SZ_X + SZ_WQ + SZ_WP + SZ_QKV);
    bf16* Vtb    = (bf16*)(ws + SZ_X + SZ_WQ + SZ_WP + 2 * SZ_QKV);
    bf16* attnbf = xbf;                                           // alias (x dead)

    const int NX = ROWS * DM;                // 8,388,608
    cvt_f32_bf16<<<(NX + 255) / 256, 256, 0, stream>>>(x, xbf, NX);
    cvt_transpose<<<(3 * DM * DM + 255) / 256, 256, 0, stream>>>(w_qkv, wqkvT, DM, 3 * DM);
    cvt_transpose<<<(DM * DM + 255) / 256, 256, 0, stream>>>(w_proj, wprojT, DM, DM);

    qkv_gemm<<<dim3(ROWS / 128, (3 * DM) / 64), 128, 0, stream>>>(xbf, wqkvT, b_qkv, Qb, Kb, Vtb);

    attn_kernel<<<BATCH * HEADS * (SEQ / 16), 32, 0, stream>>>(Qb, Kb, Vtb, attnbf);

    proj_gemm<<<dim3(ROWS / 128, DM / 64), 128, 0, stream>>>(attnbf, wprojT, b_proj, out);
}
